// TXT_TRANSFORMER_VSLT_GRUD_TOTRANSFORMER_65420941853078
// MI455X (gfx1250) — matrix operand order repaired and checked by reading, not compiled, not run on hardware
//
#include <hip/hip_runtime.h>
#include <hip/hip_bf16.h>
#include <math.h>

// ---------------------------------------------------------------------------
// gfx1250 WMMA helpers (wave32, V_WMMA_F32_16X16X32_F16)
// ---------------------------------------------------------------------------
typedef __attribute__((ext_vector_type(16))) _Float16 v16h;
typedef __attribute__((ext_vector_type(8)))  _Float16 v8h;
typedef __attribute__((ext_vector_type(8)))  float    v8f;

static __device__ __forceinline__ v8f wmma_f16(v16h a, v16h b, v8f c) {
  // (neg_a, A, neg_b, B, c_mod, C, reuse_a, reuse_b)
  return __builtin_amdgcn_wmma_f32_16x16x32_f16(false, a, false, b, (short)0, c,
                                                false, false);
}
static __device__ __forceinline__ v16h pack16(v8h lo, v8h hi) {
  return __builtin_shufflevector(lo, hi, 0, 1, 2, 3, 4, 5, 6, 7,
                                 8, 9, 10, 11, 12, 13, 14, 15);
}

// ---------------------------------------------------------------------------
// Model dimensions
// ---------------------------------------------------------------------------
#define BN   256   // batch
#define TT   24    // GRU timesteps
#define NFV  12    // features
#define DM   256   // model dim
#define HH   8     // heads (dk = 32 == WMMA K)
#define FF   512   // FFN dim
#define LTX  128   // text length
#define SFP  160   // fusion seq padded (actual 153 = 1 + 24 + 128)

// ===========================================================================
// Elementwise: xc = m*x + (1-m)*(x_d*x + (1-x_d)*x_m),  x_d = exp(-relu(d*w+b))
// ===========================================================================
__global__ void k_xc(const float* __restrict__ x, const float* __restrict__ m,
                     const float* __restrict__ d, const float* __restrict__ xm,
                     const float* __restrict__ dw, const float* __restrict__ db,
                     float* __restrict__ xc, int n) {
  int i = blockIdx.x * 256 + threadIdx.x;
  if (i >= n) return;
  int f = i % NFV;
  float xd = __expf(-fmaxf(d[i] * dw[f] + db[f], 0.0f));
  float mm = m[i];
  xc[i] = mm * x[i] + (1.0f - mm) * (xd * x[i] + (1.0f - xd) * xm[f]);
}

// ===========================================================================
// GRU-D scan: one block per batch row, thread j owns hidden unit j.
// ===========================================================================
__global__ void __launch_bounds__(256)
k_gru(const float* __restrict__ xc, const float* __restrict__ m,
      const float* __restrict__ d, const float* __restrict__ h0,
      const float* __restrict__ hdW, const float* __restrict__ hdb,
      const float* __restrict__ Wih, const float* __restrict__ Whh,
      const float* __restrict__ bih, const float* __restrict__ bhh,
      float* __restrict__ emb) {
  const int b = blockIdx.x, j = threadIdx.x;
  __shared__ float sh[DM];
  __shared__ float sin_[2 * NFV];  // [x_t | m_t]
  __shared__ float sd[NFV];
  float hc = h0[b * DM + j];
  for (int t = 0; t < TT; ++t) {
    if (j < NFV) {
      sd[j]          = d[(size_t)(b * TT + t) * NFV + j];
      sin_[j]        = xc[(size_t)(b * TT + t) * NFV + j];
      sin_[NFV + j]  = m[(size_t)(b * TT + t) * NFV + j];
    }
    __syncthreads();
    float acc = hdb[j];
    #pragma unroll
    for (int f = 0; f < NFV; ++f) acc += sd[f] * hdW[j * NFV + f];
    hc *= __expf(-fmaxf(acc, 0.0f));      // hc = h_d * hc
    sh[j] = hc;
    __syncthreads();
    float gi[3], gh[3];
    #pragma unroll
    for (int gx = 0; gx < 3; ++gx) {
      const int g = gx * DM + j;
      float a1 = bih[g];
      #pragma unroll
      for (int f = 0; f < 2 * NFV; ++f) a1 += sin_[f] * Wih[g * (2 * NFV) + f];
      float a2 = bhh[g];
      for (int k = 0; k < DM; ++k) a2 += sh[k] * Whh[(size_t)g * DM + k];
      gi[gx] = a1; gh[gx] = a2;
    }
    const float r = 1.0f / (1.0f + __expf(-(gi[0] + gh[0])));
    const float z = 1.0f / (1.0f + __expf(-(gi[1] + gh[1])));
    const float nn = tanhf(gi[2] + r * gh[2]);
    hc = (1.0f - z) * nn + z * hc;
    emb[(size_t)(b * TT + t) * DM + j] = hc;
    __syncthreads();
  }
}

// ===========================================================================
// f32 -> f16 weight pre-pack
// ===========================================================================
__global__ void k_cvt(const float* __restrict__ in, _Float16* __restrict__ out, int n) {
  int i = blockIdx.x * 256 + threadIdx.x;
  if (i < n) out[i] = (_Float16)in[i];
}

// Embedding gather (f32 table -> f16 activation). grid = B*LTX*DM/256 exact.
__global__ void k_gather(const int* __restrict__ txts, const float* __restrict__ tab,
                         _Float16* __restrict__ out) {
  const size_t idx = (size_t)blockIdx.x * 256 + threadIdx.x;
  const int j = (int)(idx & (DM - 1));
  const int tok = txts[idx >> 8];
  out[idx] = (_Float16)tab[(size_t)tok * DM + j];
}

// Fusion sequence: [cls | gru-emb(24) | txt_out(128) | zero pad(7)]  -> f16
__global__ void k_concat(const float* __restrict__ emb, const float* __restrict__ ET,
                         const float* __restrict__ fcls, _Float16* __restrict__ seq) {
  const size_t idx = (size_t)blockIdx.x * 256 + threadIdx.x;
  const int j = (int)(idx & (DM - 1));
  const size_t row = idx >> 8;
  const int s = (int)(row % SFP);
  const int b = (int)(row / SFP);
  float v;
  if (s == 0)        v = fcls[j];
  else if (s <= TT)  v = emb[((size_t)b * TT + (s - 1)) * DM + j];
  else if (s <= TT + LTX) v = ET[((size_t)b * LTX + (s - TT - 1)) * DM + j];
  else               v = 0.0f;
  seq[idx] = (_Float16)v;
}

// ===========================================================================
// WMMA GEMM:  C[M,N] = act( A[M,K](f16) @ W[N,K]^T(f16) + bias + resid )
// Register-tiled: each wave owns a 32x64 output (2 A-tiles x 4 B-tiles
// -> 8 accumulators). All 6 tile loads of a K=32 step are issued as a group
// into distinct named registers before the 8 back-to-back WMMAs, so the
// scheduler can hold them in flight and hide latency under the WMMA chain.
// Block = 4 waves stacked in M -> 128x64 block tile.
// ISA operand layouts (CDNA5 ISA §7.12.2):
//   A: lane L -> row L&15, K = (L>>4)*8 + {0..7, 16..23}
//   B: lane L -> col L&15, K = (L>>4)*8 + {0..7, 16..23}
//   C: lane L -> col L&15, row (L>>4)*8 + i
// ===========================================================================
__global__ void __launch_bounds__(128)
k_gemm(const _Float16* __restrict__ A, int lda,
       const _Float16* __restrict__ W, int ldw,
       const float* __restrict__ bias, const float* __restrict__ resid,
       float* __restrict__ Cf, _Float16* __restrict__ Ch,
       int ldc, int K, int act) {
  const int lane = threadIdx.x & 31;
  const int wave = threadIdx.x >> 5;
  const int half = lane >> 4, l15 = lane & 15;
  const int m0 = blockIdx.x * 128 + wave * 32;  // rows m0 .. m0+31
  const int n0 = blockIdx.y * 64;               // cols n0 .. n0+63

  const _Float16* Ap0 = A + (size_t)(m0 + l15) * lda + half * 8;
  const _Float16* Ap1 = Ap0 + (size_t)16 * lda;
  const _Float16* Wp0 = W + (size_t)(n0 + l15) * ldw + half * 8;
  const _Float16* Wp1 = Wp0 + (size_t)16 * ldw;
  const _Float16* Wp2 = Wp0 + (size_t)32 * ldw;
  const _Float16* Wp3 = Wp0 + (size_t)48 * ldw;

  v8f c00 = {}, c01 = {}, c02 = {}, c03 = {};
  v8f c10 = {}, c11 = {}, c12 = {}, c13 = {};
  #pragma unroll 2
  for (int kk = 0; kk < K; kk += 32) {
    __builtin_prefetch(Ap0 + kk + 128, 0, 1);
    __builtin_prefetch(Ap1 + kk + 128, 0, 1);
    __builtin_prefetch(Wp0 + kk + 128, 0, 1);
    // group all loads of this K-step into distinct live registers
    v16h a0 = pack16(*(const v8h*)(Ap0 + kk), *(const v8h*)(Ap0 + kk + 16));
    v16h a1 = pack16(*(const v8h*)(Ap1 + kk), *(const v8h*)(Ap1 + kk + 16));
    v16h b0 = pack16(*(const v8h*)(Wp0 + kk), *(const v8h*)(Wp0 + kk + 16));
    v16h b1 = pack16(*(const v8h*)(Wp1 + kk), *(const v8h*)(Wp1 + kk + 16));
    v16h b2 = pack16(*(const v8h*)(Wp2 + kk), *(const v8h*)(Wp2 + kk + 16));
    v16h b3 = pack16(*(const v8h*)(Wp3 + kk), *(const v8h*)(Wp3 + kk + 16));
    c00 = wmma_f16(a0, b0, c00);
    c10 = wmma_f16(a1, b0, c10);
    c01 = wmma_f16(a0, b1, c01);
    c11 = wmma_f16(a1, b1, c11);
    c02 = wmma_f16(a0, b2, c02);
    c12 = wmma_f16(a1, b2, c12);
    c03 = wmma_f16(a0, b3, c03);
    c13 = wmma_f16(a1, b3, c13);
  }

  v8f acc[2][4] = {{c00, c01, c02, c03}, {c10, c11, c12, c13}};
  #pragma unroll
  for (int nt = 0; nt < 4; ++nt) {
    const int cn = n0 + nt * 16 + l15;
    const float bb = bias ? bias[cn] : 0.0f;
    #pragma unroll
    for (int mt = 0; mt < 2; ++mt) {
      #pragma unroll
      for (int i = 0; i < 8; ++i) {
        const int cm = m0 + mt * 16 + half * 8 + i;
        float v = acc[mt][nt][i] + bb;
        if (resid) v += resid[(size_t)cm * ldc + cn];
        if (act)   v = fmaxf(v, 0.0f);
        if (Cf) Cf[(size_t)cm * ldc + cn] = v;
        if (Ch) Ch[(size_t)cm * ldc + cn] = (_Float16)v;
      }
    }
  }
}

// ===========================================================================
// LayerNorm row kernel: y = (x-mu)*rsqrt(var+eps)*g + b [+PE] [* npad]
// one block (256 threads) per row; writes f32 master + f16 copy.
// ===========================================================================
__global__ void __launch_bounds__(256)
k_ln(const float* __restrict__ X, float* __restrict__ Y, _Float16* __restrict__ Yh,
     const float* __restrict__ g, const float* __restrict__ bta,
     int S, int addPE, const int* __restrict__ lens, int lenoff) {
  const int row = blockIdx.x;
  const int j = threadIdx.x;
  __shared__ float red[256];
  const float xv = X[(size_t)row * DM + j];
  red[j] = xv; __syncthreads();
  #pragma unroll
  for (int s = 128; s > 0; s >>= 1) { if (j < s) red[j] += red[j + s]; __syncthreads(); }
  const float mu = red[0] * (1.0f / 256.0f);
  __syncthreads();
  const float dx = xv - mu;
  red[j] = dx * dx; __syncthreads();
  #pragma unroll
  for (int s = 128; s > 0; s >>= 1) { if (j < s) red[j] += red[j + s]; __syncthreads(); }
  const float var = red[0] * (1.0f / 256.0f);
  float y = dx * rsqrtf(var + 1e-5f) * g[j] + bta[j];
  const int sp = row % S;
  if (addPE) {  // sinusoidal PE computed on the fly
    const float ex  = (float)(j & ~1) * (1.0f / 256.0f);
    const float ang = (float)sp * __powf(10000.0f, -ex);
    y += (j & 1) ? __cosf(ang) : __sinf(ang);
  }
  if (lens && sp >= lens[row / S] + lenoff) y = 0.0f;  // npad mask
  Y[(size_t)row * DM + j] = y;
  if (Yh) Yh[(size_t)row * DM + j] = (_Float16)y;
}

// ===========================================================================
// Fused attention: one wave per (batch, head, 16-query tile).
// Q/K/V live in one fused QKV buffer (row stride ld = 768).
// scores = Q@K^T/sqrt(32) (WMMA, dk=32 == one K step), mask, softmax in LDS,
// ctx = P@V (WMMA over S/32 K steps, two dk column tiles).
// ===========================================================================
__global__ void __launch_bounds__(32)
k_attn(const _Float16* __restrict__ Q, const _Float16* __restrict__ Km,
       const _Float16* __restrict__ V, _Float16* __restrict__ C,
       int S, int ld, const int* __restrict__ lens, int lenoff) {
  __shared__ float sc[16][SFP];
  const int lane = threadIdx.x;
  const int half = lane >> 4, l15 = lane & 15;
  const int b = blockIdx.z, h = blockIdx.y, q0 = blockIdx.x * 16;
  const int len = lens ? (lens[b] + lenoff) : S;

  // Q tile (A layout), K-dim = dk(=32)
  v16h aq;
  {
    const _Float16* Qp = Q + (size_t)(b * S + q0 + l15) * ld + h * 32 + half * 8;
    aq = pack16(*(const v8h*)Qp, *(const v8h*)(Qp + 16));
  }
  const float scale = 0.17677669529663687f;  // 1/sqrt(32)
  for (int k0 = 0; k0 < S; k0 += 16) {
    const _Float16* Kp = Km + (size_t)(b * S + k0 + l15) * ld + h * 32 + half * 8;
    v16h bk = pack16(*(const v8h*)Kp, *(const v8h*)(Kp + 16));        // B layout: col = key, K = dk
    v8f c = {};
    c = wmma_f16(aq, bk, c);
    const int key = k0 + l15;
    #pragma unroll
    for (int i = 0; i < 8; ++i) {
      float v = c[i] * scale;
      if (key >= len) v = -1.0e9f;
      sc[half * 8 + i][key] = v;
    }
  }
  __syncthreads();
  if (lane < 16) {  // softmax, one lane per query row
    float mx = -3.0e38f;
    for (int s2 = 0; s2 < S; ++s2) mx = fmaxf(mx, sc[lane][s2]);
    float sum = 0.0f;
    for (int s2 = 0; s2 < S; ++s2) { float e = __expf(sc[lane][s2] - mx); sc[lane][s2] = e; sum += e; }
    const float inv = 1.0f / sum;
    for (int s2 = 0; s2 < S; ++s2) sc[lane][s2] *= inv;
  }
  __syncthreads();
  #pragma unroll
  for (int nt = 0; nt < 2; ++nt) {  // dk = 32 -> two 16-col output tiles
    v8f acc = {};
    for (int ck = 0; ck < S; ck += 32) {
      v16h ap;
      #pragma unroll
      for (int i = 0; i < 8; ++i) {  // A layout from LDS probs
        ap[i]     = (_Float16)sc[l15][ck + half * 16 + i];
        ap[i + 8] = (_Float16)sc[l15][ck + half * 16 + 8 + i];
      }
      const _Float16* Vp = V + (size_t)(b * S + ck + half * 16) * ld + h * 32 + nt * 16 + l15;
      v16h bv;
      #pragma unroll
      for (int i = 0; i < 16; ++i) bv[i] = Vp[(size_t)i * ld];  // B: K = seq idx
      acc = wmma_f16(ap, bv, acc);
    }
    #pragma unroll
    for (int i = 0; i < 8; ++i)
      C[(size_t)(b * S + q0 + half * 8 + i) * DM + h * 32 + nt * 16 + l15] = (_Float16)acc[i];
  }
}

// ===========================================================================
// Heads: ci = [cls(256), age, gen];  12 x (258->64 relu-bn -> 1) -> sigmoid
// ===========================================================================
__global__ void __launch_bounds__(64)
k_heads(const float* __restrict__ EF, const float* __restrict__ age,
        const float* __restrict__ gen, const float* __restrict__ fc1W,
        const float* __restrict__ fc1b, const float* __restrict__ bng,
        const float* __restrict__ bnb, const float* __restrict__ fc2W,
        const float* __restrict__ fc2b, float* __restrict__ out, int S) {
  const int kh = blockIdx.x, b = blockIdx.y, t = threadIdx.x;
  __shared__ float ci[DM + 2];
  __shared__ float h1[64];
  for (int i = t; i < DM; i += 64) ci[i] = EF[(size_t)(b * S) * DM + i];  // cls row
  if (t == 0) { ci[DM] = age[b]; ci[DM + 1] = gen[b]; }
  __syncthreads();
  float a = fc1b[kh * 64 + t];
  const float* w = fc1W + ((size_t)kh * 64 + t) * (DM + 2);
  for (int j = 0; j < DM + 2; ++j) a += ci[j] * w[j];
  a = a * rsqrtf(1.0f + 1e-5f) * bng[kh * 64 + t] + bnb[kh * 64 + t];
  h1[t] = fmaxf(a, 0.0f);
  __syncthreads();
  if (t == 0) {
    float s = fc2b[kh];
    for (int j = 0; j < 64; ++j) s += h1[j] * fc2W[kh * 64 + j];
    out[(size_t)kh * gridDim.y + b] = 1.0f / (1.0f + __expf(-s));
  }
}

// ===========================================================================
// Host driver
// ===========================================================================
extern "C" void kernel_launch(void* const* d_in, const int* in_sizes, int n_in,
                              void* d_out, int out_size, void* d_ws, size_t ws_size,
                              hipStream_t stream) {
  (void)in_sizes; (void)n_in; (void)out_size; (void)ws_size;

  const float* x       = (const float*)d_in[0];
  const float* h0      = (const float*)d_in[1];
  const float* m       = (const float*)d_in[2];
  const float* dmat    = (const float*)d_in[3];
  const float* x_m     = (const float*)d_in[4];
  const float* age     = (const float*)d_in[5];
  const float* gen     = (const float*)d_in[6];
  const int*   txts    = (const int*)d_in[8];
  const int*   txt_len = (const int*)d_in[9];
  const float* dec_w   = (const float*)d_in[10];
  const float* dec_b   = (const float*)d_in[11];
  const float* hd_W    = (const float*)d_in[12];
  const float* hd_b    = (const float*)d_in[13];
  const float* Wih     = (const float*)d_in[14];
  const float* Whh     = (const float*)d_in[15];
  const float* bih     = (const float*)d_in[16];
  const float* bhh     = (const float*)d_in[17];
  const float* txt_emb = (const float*)d_in[18];
  const float* enc_p[2][10];  // [0]=t_, [1]=f_
  for (int e = 0; e < 2; ++e)
    for (int k = 0; k < 10; ++k) enc_p[e][k] = (const float*)d_in[19 + e * 10 + k];
  const float* f_cls = (const float*)d_in[39];
  const float* fc1W  = (const float*)d_in[40];
  const float* fc1b  = (const float*)d_in[41];
  const float* bng   = (const float*)d_in[42];
  const float* bnb   = (const float*)d_in[43];
  const float* fc2W  = (const float*)d_in[44];
  const float* fc2b  = (const float*)d_in[45];

  // ---- workspace carve (all 256B aligned) ----
  char* base = (char*)d_ws;
  size_t off = 0;
  auto carve = [&](size_t bytes) -> void* {
    void* p = base + off;
    off = (off + bytes + 255) & ~(size_t)255;
    return p;
  };
  float*     XC    = (float*)carve((size_t)BN * TT * NFV * 4);
  float*     EMB   = (float*)carve((size_t)BN * TT * DM * 4);
  float*     ET    = (float*)carve((size_t)BN * LTX * DM * 4);   // text enc f32
  float*     EF    = (float*)carve((size_t)BN * SFP * DM * 4);   // fusion enc f32
  float*     TMP   = (float*)carve((size_t)BN * SFP * DM * 4);   // pre-LN buffer
  _Float16*  XG16  = (_Float16*)carve((size_t)BN * LTX * DM * 2);
  _Float16*  SQ16  = (_Float16*)carve((size_t)BN * SFP * DM * 2);
  _Float16*  ET16  = (_Float16*)carve((size_t)BN * LTX * DM * 2);
  _Float16*  EF16  = (_Float16*)carve((size_t)BN * SFP * DM * 2);
  _Float16*  QKV16 = (_Float16*)carve((size_t)BN * SFP * 3 * DM * 2);  // fused q|k|v
  _Float16*  CT16  = (_Float16*)carve((size_t)BN * SFP * DM * 2);
  _Float16*  F116  = (_Float16*)carve((size_t)BN * SFP * FF * 2);
  _Float16*  W16[2][4];  // per encoder: inW, attnW(2*4), ffW1, ffW2 packed f16
  const size_t wsz[4] = { (size_t)DM * DM, (size_t)2 * 4 * DM * DM,
                          (size_t)2 * FF * DM, (size_t)2 * DM * FF };
  for (int e = 0; e < 2; ++e)
    for (int k = 0; k < 4; ++k) W16[e][k] = (_Float16*)carve(wsz[k] * 2);

  auto cvt = [&](const float* in, _Float16* o, size_t n) {
    k_cvt<<<(unsigned)((n + 255) / 256), 256, 0, stream>>>(in, o, (int)n);
  };
  // pre-pack weights to f16: in_W, attn_W, ff_W1, ff_W2 for both encoders
  const int widx[4] = { 0, 2, 4, 6 };  // offsets inside enc_p
  for (int e = 0; e < 2; ++e)
    for (int k = 0; k < 4; ++k) cvt(enc_p[e][widx[k]], W16[e][k], wsz[k]);

  auto gemm = [&](const _Float16* A, int lda, const _Float16* W, int ldw,
                  const float* bias, const float* resid,
                  float* Cf, _Float16* Ch, int ldc, int M, int N, int K, int act) {
    k_gemm<<<dim3(M / 128, N / 64), dim3(128), 0, stream>>>(
        A, lda, W, ldw, bias, resid, Cf, Ch, ldc, K, act);
  };

  // ---- GRU-D branch ----
  k_xc<<<(BN * TT * NFV + 255) / 256, 256, 0, stream>>>(x, m, dmat, x_m, dec_w,
                                                        dec_b, XC, BN * TT * NFV);
  k_gru<<<BN, 256, 0, stream>>>(XC, m, dmat, h0, hd_W, hd_b, Wih, Whh, bih, bhh, EMB);

  // ---- text embedding gather ----
  k_gather<<<(BN * LTX * DM) / 256, 256, 0, stream>>>(txts, txt_emb, XG16);

  // ---- generic encoder ----
  auto encoder = [&](int e, int S, const _Float16* X16, float* E, _Float16* E16,
                     const int* lens, int lenoff) {
    const int M = BN * S;
    const float* inb = enc_p[e][1];
    const float* ab  = enc_p[e][3];
    const float* fb1 = enc_p[e][5];
    const float* fb2 = enc_p[e][7];
    const float* lng = enc_p[e][8];
    const float* lnb = enc_p[e][9];
    // input projection + LN + PE
    gemm(X16, DM, W16[e][0], DM, inb, nullptr, TMP, nullptr, DM, M, DM, DM, 0);
    k_ln<<<M, 256, 0, stream>>>(TMP, E, E16, lng, lnb, S, 1, nullptr, 0);
    for (int l = 0; l < 2; ++l) {
      // fused QKV projection: rows q(0..255)|k(256..511)|v(512..767) contiguous
      gemm(E16, DM, W16[e][1] + (size_t)(l * 4) * DM * DM, DM,
           ab + (size_t)(l * 4) * DM, nullptr, nullptr, QKV16, 3 * DM,
           M, 3 * DM, DM, 0);
      k_attn<<<dim3(S / 16, HH, BN), 32, 0, stream>>>(
          QKV16, QKV16 + DM, QKV16 + 2 * DM, CT16, S, 3 * DM, lens, lenoff);
      gemm(CT16, DM, W16[e][1] + (size_t)(l * 4 + 3) * DM * DM, DM,
           ab + (size_t)(l * 4 + 3) * DM, E, TMP, nullptr, DM, M, DM, DM, 0);
      k_ln<<<M, 256, 0, stream>>>(TMP, E, E16, lng + (size_t)(1 + 2 * l) * DM,
                                  lnb + (size_t)(1 + 2 * l) * DM, S, 0, lens, lenoff);
      gemm(E16, DM, W16[e][2] + (size_t)l * FF * DM, DM, fb1 + (size_t)l * FF,
           nullptr, nullptr, F116, FF, M, FF, DM, 1);  // FFN1 + ReLU
      gemm(F116, FF, W16[e][3] + (size_t)l * DM * FF, FF, fb2 + (size_t)l * DM,
           E, TMP, nullptr, DM, M, DM, FF, 0);         // FFN2 + residual
      k_ln<<<M, 256, 0, stream>>>(TMP, E, E16, lng + (size_t)(2 + 2 * l) * DM,
                                  lnb + (size_t)(2 + 2 * l) * DM, S, 0, lens, lenoff);
    }
  };

  // text encoder (S = 128, key mask = txt_lengths)
  encoder(0, LTX, XG16, ET, ET16, txt_len, 0);

  // fusion sequence + encoder (S padded 160, fin_len = txt_len + 25)
  k_concat<<<(BN * SFP * DM) / 256, 256, 0, stream>>>(EMB, ET, f_cls, SQ16);
  encoder(1, SFP, SQ16, EF, EF16, txt_len, TT + 1);

  // heads -> (12, B, 1) sigmoid
  k_heads<<<dim3(12, BN), 64, 0, stream>>>(EF, age, gen, fc1W, fc1b, bng, bnb,
                                           fc2W, fc2b, (float*)d_out, SFP);
}
